// Decoder_82154134438071
// MI455X (gfx1250) — compile-verified, matrix operand order checked by reading, not run on hardware
//
#include <hip/hip_runtime.h>
#include <hip/hip_bf16.h>
#include <math.h>

// ---------------------------------------------------------------------------
// S=128, B=32, H=512, E=512, L=2, V=32000.  d_out: [B,128,V] fp32.
// bf16 WMMA (v_wmma_f32_16x16x32_bf16) for all GEMMs, fp32 accumulate.
// Each wave owns 2 M-tiles (M=32) so every W tile is loaded exactly once.
// A-panel (32x512 bf16 = 32KB) staged to LDS per block via the CDNA5
// async-to-LDS path (ASYNCcnt) when the builtin is available.
// ---------------------------------------------------------------------------

#define BSZ   32
#define SEQ   128
#define HID   512
#define EMB   512
#define NLAY  2
#define VOC   32000
#define NH    8
#define DK    64

typedef __attribute__((ext_vector_type(16))) __bf16 v16bf;
typedef __attribute__((ext_vector_type(8)))  __bf16 v8bf;
typedef __attribute__((ext_vector_type(8)))  float  v8f;
typedef int i128_t __attribute__((vector_size(16)));   // matches builtin param

#if __has_builtin(__builtin_amdgcn_global_load_async_to_lds_b128) && \
    __has_builtin(__builtin_amdgcn_s_wait_asynccnt)
#define ASYNC_LDS_COPY 1
#else
#define ASYNC_LDS_COPY 0
#endif

union BfTileU { v16bf v; v8bf h[2]; };

// 16x32 bf16 tile operand in the CDNA5 WMMA A/B register layout:
//   lanes 0-15 : row = lane,    K chunks {k..k+7, k+16..k+23}
//   lanes 16-31: row = lane-16, K chunks {k+8..k+15, k+24..k+31}
__device__ __forceinline__ v16bf wmma_load_tile(const __bf16* base, int row0,
                                                int ld, int kbase, int lane) {
    const __bf16* p = base + (size_t)(row0 + (lane & 15)) * (size_t)ld
                           + (size_t)(kbase + ((lane >> 4) << 3));
    BfTileU t;
    t.h[0] = *(const v8bf*)(p);
    t.h[1] = *(const v8bf*)(p + 16);
    return t.v;
}

// Cooperative copy of a contiguous panel (16B chunks) into LDS.
__device__ __forceinline__ void stage_panel(__bf16* sA, const __bf16* A,
                                            int nchunks) {
    int tid = (int)threadIdx.x, nt = (int)blockDim.x;
    for (int i = tid; i < nchunks; i += nt) {
#if ASYNC_LDS_COPY
        __builtin_amdgcn_global_load_async_to_lds_b128(
            (__attribute__((address_space(1))) i128_t*)((const char*)A + (size_t)i * 16),
            (__attribute__((address_space(3))) i128_t*)((char*)sA + (size_t)i * 16),
            0, 0);
#else
        *(v8bf*)((char*)sA + (size_t)i * 16) =
            *(const v8bf*)((const char*)A + (size_t)i * 16);
#endif
    }
#if ASYNC_LDS_COPY
    __builtin_amdgcn_s_wait_asynccnt(0);
#endif
    __syncthreads();
}

// GEMM: Y[M,N] = A[M,K](bf16) * W[N,K]^T(bf16) + bias[N] (+Y if addY).
// One wave computes a 32x16 tile (2 m-tiles); B tile feeds 2 WMMAs.
// STAGE: copy the whole A panel (lda must equal K, M*K*2 <= smem) into LDS.
template <bool STAGE>
__global__ void Decoder_gemm_wmma(const __bf16* __restrict__ A, int lda,
                                  const __bf16* __restrict__ W, int ldw,
                                  const float* __restrict__ bias,
                                  float* __restrict__ Y, size_t ldy,
                                  __bf16* __restrict__ Ybf, int ldybf,
                                  int M, int N, int K, int addY) {
    extern __shared__ __bf16 sA[];
    const __bf16* Aeff = A;
    if (STAGE) {                       // all threads participate before guard
        stage_panel(sA, A, (M * K) >> 3);
        Aeff = sA;
    }

    int wid  = (int)((blockIdx.x * blockDim.x + threadIdx.x) >> 5);
    int lane = (int)(threadIdx.x & 31);
    int ntiles = N >> 4;
    if (wid >= (M >> 5) * ntiles) return;   // whole-wave uniform exit
    int mg = wid / ntiles, nt = wid % ntiles;
    int m0 = mg << 5, n0 = nt << 4;

    // C/D layout: lane<16 -> n=n0+lane, rows mb+r ; lane>=16 -> n=n0+lane-16,
    // rows mb+8+r  (r = vector element 0..7)
    int n = n0 + (lane & 15);
    int mbase = m0 + ((lane >> 4) << 3);

    v8f acc[2];
    float bv = bias ? bias[n] : 0.0f;
#pragma unroll
    for (int mt = 0; mt < 2; ++mt) {
#pragma unroll
        for (int r = 0; r < 8; ++r) acc[mt][r] = bv;
        if (addY) {
#pragma unroll
            for (int r = 0; r < 8; ++r)
                acc[mt][r] += Y[(size_t)(mbase + mt * 16 + r) * ldy + n];
        }
    }

    const __bf16* wrow = W + (size_t)(n0 + (lane & 15)) * (size_t)ldw
                           + ((lane >> 4) << 3);
#pragma unroll 4
    for (int k = 0; k < K; k += 32) {
        if (k + 128 < K) __builtin_prefetch(wrow + k + 128, 0, 1);
        v16bf b  = wmma_load_tile(W, n0, ldw, k, lane);
        v16bf a0 = wmma_load_tile(Aeff, m0,      lda, k, lane);
        v16bf a1 = wmma_load_tile(Aeff, m0 + 16, lda, k, lane);
        acc[0] = __builtin_amdgcn_wmma_f32_16x16x32_bf16(
                     false, a0, false, b, (short)0, acc[0], false, false);
        acc[1] = __builtin_amdgcn_wmma_f32_16x16x32_bf16(
                     false, a1, false, b, (short)0, acc[1], false, false);
    }

#pragma unroll
    for (int mt = 0; mt < 2; ++mt) {
#pragma unroll
        for (int r = 0; r < 8; ++r)
            Y[(size_t)(mbase + mt * 16 + r) * ldy + n] = acc[mt][r];
    }
    if (Ybf) {
#pragma unroll
        for (int mt = 0; mt < 2; ++mt)
#pragma unroll
            for (int r = 0; r < 8; ++r)
                Ybf[(mbase + mt * 16 + r) * ldybf + n] = (__bf16)acc[mt][r];
    }
}

__global__ void Decoder_f32_to_bf16(const float* __restrict__ src,
                                    __bf16* __restrict__ dst, int n) {
    int i = blockIdx.x * blockDim.x + threadIdx.x;
    if (i < n) dst[i] = (__bf16)src[i];
}

__global__ void Decoder_init(const float* __restrict__ hidden,
                             const float* __restrict__ cell,
                             __bf16* __restrict__ hbf,
                             float* __restrict__ c, int* __restrict__ tok) {
    int i = blockIdx.x * blockDim.x + threadIdx.x;
    if (i < NLAY * BSZ * HID) {
        hbf[i] = (__bf16)hidden[i];
        c[i]   = cell[i];
    }
    if (i < BSZ) tok[i] = 0;   // SOS
}

__global__ void Decoder_zero_t0(float* __restrict__ out) {
    int i = blockIdx.x * blockDim.x + threadIdx.x;
    if (i >= BSZ * VOC) return;
    int b = i / VOC, v = i % VOC;
    out[(size_t)b * SEQ * VOC + v] = 0.0f;
}

__global__ void Decoder_embed(const int* __restrict__ tok,
                              const float* __restrict__ emb,
                              __bf16* __restrict__ xbf) {
    int i = blockIdx.x * blockDim.x + threadIdx.x;
    if (i >= BSZ * EMB) return;
    int b = i >> 9, e = i & 511;
    xbf[i] = (__bf16)emb[(size_t)tok[b] * EMB + e];
}

// gates[b, 0:2048] in order i,f,g,o -> update c, emit h as bf16
__global__ void Decoder_lstm_update(const float* __restrict__ gates,
                                    float* __restrict__ c,
                                    __bf16* __restrict__ hbf) {
    int i = blockIdx.x * blockDim.x + threadIdx.x;
    if (i >= BSZ * HID) return;
    int b = i >> 9, j = i & 511;
    const float* g = gates + (size_t)b * (4 * HID);
    float gi = g[j], gf = g[HID + j], gg = g[2 * HID + j], go = g[3 * HID + j];
    float si = 1.0f / (1.0f + __expf(-gi));
    float sf = 1.0f / (1.0f + __expf(-gf));
    float so = 1.0f / (1.0f + __expf(-go));
    float cn = sf * c[i] + si * tanhf(gg);
    c[i]   = cn;
    hbf[i] = (__bf16)(so * tanhf(cn));
}

// One wave per (batch, head). kc/vc layout: [s*B + b][HID] fp32.
__global__ void Decoder_attention(const float* __restrict__ q,
                                  const float* __restrict__ kc,
                                  const float* __restrict__ vc,
                                  __bf16* __restrict__ attn_bf) {
    int wid  = (int)((blockIdx.x * blockDim.x + threadIdx.x) >> 5);
    int lane = (int)(threadIdx.x & 31);
    if (wid >= BSZ * NH) return;
    int b = wid >> 3, h = wid & 7;
    const float* qp = q + (size_t)b * HID + h * DK;
    float q0 = qp[lane], q1 = qp[lane + 32];
    const float scale = 0.125f;   // 1/sqrt(64)

    float scores[4];
#pragma unroll
    for (int j = 0; j < 4; ++j) {
        float sc = 0.0f;
        for (int si = 0; si < 32; ++si) {
            int s = j * 32 + si;
            const float* kp = kc + (size_t)(s * BSZ + b) * HID + h * DK;
            float p = q0 * kp[lane] + q1 * kp[lane + 32];
#pragma unroll
            for (int m = 16; m >= 1; m >>= 1) p += __shfl_xor(p, m, 32);
            if (si == lane) sc = p * scale;
        }
        scores[j] = sc;
    }
    float mx = fmaxf(fmaxf(scores[0], scores[1]), fmaxf(scores[2], scores[3]));
#pragma unroll
    for (int m = 16; m >= 1; m >>= 1) mx = fmaxf(mx, __shfl_xor(mx, m, 32));
    float pr[4], sum = 0.0f;
#pragma unroll
    for (int j = 0; j < 4; ++j) { pr[j] = __expf(scores[j] - mx); sum += pr[j]; }
#pragma unroll
    for (int m = 16; m >= 1; m >>= 1) sum += __shfl_xor(sum, m, 32);
    float inv = 1.0f / sum;

    float a0 = 0.0f, a1 = 0.0f;
#pragma unroll
    for (int j = 0; j < 4; ++j) {
        for (int si = 0; si < 32; ++si) {
            float ps = __shfl(pr[j], si, 32);
            int s = j * 32 + si;
            const float* vp = vc + (size_t)(s * BSZ + b) * HID + h * DK;
            a0 += ps * vp[lane];
            a1 += ps * vp[lane + 32];
        }
    }
    attn_bf[(size_t)b * HID + h * DK + lane]      = (__bf16)(a0 * inv);
    attn_bf[(size_t)b * HID + h * DK + lane + 32] = (__bf16)(a1 * inv);
}

// argmax over V per batch row; first-occurrence tie-break (matches jnp.argmax)
__global__ void Decoder_argmax(const float* __restrict__ logits, size_t ld,
                               int* __restrict__ tok) {
    __shared__ float sv[256];
    __shared__ int   si_[256];
    int b = blockIdx.x, t = threadIdx.x;
    const float* row = logits + (size_t)b * ld;
    float best = -INFINITY; int bi = 0;
    for (int v = t; v < VOC; v += 256) {
        float x = row[v];
        if (x > best) { best = x; bi = v; }   // v increasing -> first max kept
    }
    sv[t] = best; si_[t] = bi;
    __syncthreads();
    for (int off = 128; off > 0; off >>= 1) {
        if (t < off) {
            if (sv[t + off] > sv[t] ||
                (sv[t + off] == sv[t] && si_[t + off] < si_[t])) {
                sv[t] = sv[t + off]; si_[t] = si_[t + off];
            }
        }
        __syncthreads();
    }
    if (t == 0) tok[b] = si_[0];
}

// ---------------------------------------------------------------------------

extern "C" void kernel_launch(void* const* d_in, const int* in_sizes, int n_in,
                              void* d_out, int out_size, void* d_ws, size_t ws_size,
                              hipStream_t stream) {
    const float* hs     = (const float*)d_in[0];
    const float* hidden = (const float*)d_in[1];
    const float* cell   = (const float*)d_in[2];
    const float* emb    = (const float*)d_in[3];
    const float* W_ih   = (const float*)d_in[4];
    const float* W_hh   = (const float*)d_in[5];
    const float* b_ih   = (const float*)d_in[6];
    const float* b_hh   = (const float*)d_in[7];
    const float* Wq     = (const float*)d_in[8];
    const float* bq     = (const float*)d_in[9];
    const float* Wk     = (const float*)d_in[10];
    const float* bk     = (const float*)d_in[11];
    const float* Wv     = (const float*)d_in[12];
    const float* bv     = (const float*)d_in[13];
    const float* Wo     = (const float*)d_in[14];
    const float* bo     = (const float*)d_in[15];
    const float* W_out  = (const float*)d_in[16];
    const float* b_out  = (const float*)d_in[17];
    float* out = (float*)d_out;

    // ---- workspace carve-up (~66 MB) ----
    char*  wsb = (char*)d_ws;
    size_t off = 0;
    auto alloc = [&](size_t bytes) -> char* {
        char* p = wsb + off;
        off += (bytes + 255) & ~(size_t)255;
        return p;
    };
    __bf16* hs_bf   = (__bf16*)alloc((size_t)SEQ * BSZ * HID * 2);
    __bf16* Wih_bf  = (__bf16*)alloc((size_t)NLAY * 4 * HID * EMB * 2);
    __bf16* Whh_bf  = (__bf16*)alloc((size_t)NLAY * 4 * HID * HID * 2);
    __bf16* Wq_bf   = (__bf16*)alloc((size_t)HID * HID * 2);
    __bf16* Wk_bf   = (__bf16*)alloc((size_t)HID * HID * 2);
    __bf16* Wv_bf   = (__bf16*)alloc((size_t)HID * HID * 2);
    __bf16* Wo_bf   = (__bf16*)alloc((size_t)HID * HID * 2);
    __bf16* Wout_bf = (__bf16*)alloc((size_t)VOC * HID * 2);
    float*  kcache  = (float*)alloc((size_t)SEQ * BSZ * HID * 4);
    float*  vcache  = (float*)alloc((size_t)SEQ * BSZ * HID * 4);
    float*  gates   = (float*)alloc((size_t)BSZ * 4 * HID * 4);
    float*  cstate  = (float*)alloc((size_t)NLAY * BSZ * HID * 4);
    __bf16* h_bf    = (__bf16*)alloc((size_t)NLAY * BSZ * HID * 2);
    __bf16* x_bf    = (__bf16*)alloc((size_t)BSZ * EMB * 2);
    float*  qbuf    = (float*)alloc((size_t)BSZ * HID * 4);
    __bf16* attn_bf = (__bf16*)alloc((size_t)BSZ * HID * 2);
    float*  ao_f    = (float*)alloc((size_t)BSZ * HID * 4);
    __bf16* ao_bf   = (__bf16*)alloc((size_t)BSZ * HID * 2);
    int*    tok     = (int*)alloc((size_t)BSZ * 4);
    (void)ws_size; (void)in_sizes; (void)n_in; (void)out_size;

    const int SMEM = BSZ * HID * 2;   // 32KB staged A panel

    auto conv = [&](const float* s, __bf16* d, int n) {
        Decoder_f32_to_bf16<<<(n + 255) / 256, 256, 0, stream>>>(s, d, n);
    };
    // Staged GEMM (M == 32, lda == K): A panel in LDS.
    auto gemmS = [&](const __bf16* A, const __bf16* W, int ldw,
                     const float* bias, float* Y, size_t ldy,
                     __bf16* Ybf, int ldybf, int N, int K, int addY) {
        int waves  = (32 / 32) * (N / 16);
        int blocks = (waves + 3) / 4;             // 4 waves (128 thr) / block
        Decoder_gemm_wmma<true><<<blocks, 128, SMEM, stream>>>(
            A, K, W, ldw, bias, Y, ldy, Ybf, ldybf, 32, N, K, addY);
    };
    // Unstaged GEMM (large M), A from global.
    auto gemmL = [&](const __bf16* A, int lda, const __bf16* W, int ldw,
                     const float* bias, float* Y, size_t ldy,
                     int M, int N, int K) {
        int waves  = (M / 32) * (N / 16);
        int blocks = (waves + 3) / 4;
        Decoder_gemm_wmma<false><<<blocks, 128, 0, stream>>>(
            A, lda, W, ldw, bias, Y, ldy, nullptr, 0, M, N, K, 0);
    };

    // ---- one-time prep: bf16 weights, state init, t=0 plane ----
    conv(hs,    hs_bf,   SEQ * BSZ * HID);
    conv(W_ih,  Wih_bf,  NLAY * 4 * HID * EMB);
    conv(W_hh,  Whh_bf,  NLAY * 4 * HID * HID);
    conv(Wq,    Wq_bf,   HID * HID);
    conv(Wk,    Wk_bf,   HID * HID);
    conv(Wv,    Wv_bf,   HID * HID);
    conv(Wo,    Wo_bf,   HID * HID);
    conv(W_out, Wout_bf, VOC * HID);
    Decoder_init<<<(NLAY * BSZ * HID + 255) / 256, 256, 0, stream>>>(
        hidden, cell, h_bf, cstate, tok);
    Decoder_zero_t0<<<(BSZ * VOC + 255) / 256, 256, 0, stream>>>(out);

    // K/V caches: [S*B, H] = hs_bf [S*B, H] @ Wk/Wv^T + bias
    gemmL(hs_bf, HID, Wk_bf, HID, bk, kcache, HID, SEQ * BSZ, HID, HID);
    gemmL(hs_bf, HID, Wv_bf, HID, bv, vcache, HID, SEQ * BSZ, HID, HID);

    // ---- autoregressive decode: 127 steps ----
    for (int t = 1; t < SEQ; ++t) {
        Decoder_embed<<<(BSZ * EMB + 255) / 256, 256, 0, stream>>>(tok, emb, x_bf);

        // LSTM layer 0
        gemmS(x_bf, Wih_bf, EMB, b_ih, gates, 4 * HID, nullptr, 0,
              4 * HID, EMB, 0);
        gemmS(h_bf, Whh_bf, HID, b_hh, gates, 4 * HID, nullptr, 0,
              4 * HID, HID, 1);
        Decoder_lstm_update<<<(BSZ * HID + 255) / 256, 256, 0, stream>>>(
            gates, cstate, h_bf);

        // LSTM layer 1 (input = new h0)
        gemmS(h_bf, Wih_bf + (size_t)4 * HID * EMB, EMB, b_ih + 4 * HID,
              gates, 4 * HID, nullptr, 0, 4 * HID, EMB, 0);
        gemmS(h_bf + (size_t)BSZ * HID, Whh_bf + (size_t)4 * HID * HID, HID,
              b_hh + 4 * HID, gates, 4 * HID, nullptr, 0, 4 * HID, HID, 1);
        Decoder_lstm_update<<<(BSZ * HID + 255) / 256, 256, 0, stream>>>(
            gates, cstate + (size_t)BSZ * HID, h_bf + (size_t)BSZ * HID);

        // q = h1 @ Wq^T + bq
        gemmS(h_bf + (size_t)BSZ * HID, Wq_bf, HID, bq, qbuf, HID,
              nullptr, 0, HID, HID, 0);
        // softmax attention over cached K/V
        Decoder_attention<<<64, 128, 0, stream>>>(qbuf, kcache, vcache, attn_bf);
        // attn_out = attn @ Wo^T + bo  (bf16 mirror feeds logits GEMM)
        gemmS(attn_bf, Wo_bf, HID, bo, ao_f, HID, ao_bf, HID, HID, HID, 0);
        // logits -> directly into d_out[b][t][:]  (row b stride = 128*V)
        gemmS(ao_bf, Wout_bf, HID, b_out, out + (size_t)t * VOC,
              (size_t)SEQ * VOC, nullptr, 0, VOC, HID, 0);
        // next token
        Decoder_argmax<<<BSZ, 256, 0, stream>>>(out + (size_t)t * VOC,
                                                (size_t)SEQ * VOC, tok);
    }
}
